// HardNegativeContrastiveLoss_56573309224206
// MI455X (gfx1250) — matrix operand order repaired and checked by reading, not compiled, not run on hardware
//
#include <hip/hip_runtime.h>
#include <hip/hip_bf16.h>

typedef __attribute__((ext_vector_type(16))) _Float16 v16h;
typedef __attribute__((ext_vector_type(8)))  _Float16 v8h;
typedef __attribute__((ext_vector_type(4)))  _Float16 v4h;
typedef __attribute__((ext_vector_type(8)))  float    v8f;
typedef __attribute__((ext_vector_type(4)))  unsigned int v4u;
typedef __attribute__((ext_vector_type(8)))  int      v8i;
typedef __attribute__((ext_vector_type(4)))  int      v4i;

#define DDIM         128
#define TEMP_INV     10.0f
#define MARGIN_C     0.5f
#define PANEL_COLS   64
#define COL_STRIDE_H 136                 // 128 halfs + 8 halfs TDM pad
#define COL_STRIDE_B 272                 // bytes; 4n mod 64 bank groups -> conflict-free
#define PANEL_BYTES  (PANEL_COLS * COL_STRIDE_B)

__device__ __forceinline__ float waveReduceSum32(float v) {
#pragma unroll
    for (int m = 16; m >= 1; m >>= 1) v += __shfl_xor(v, m, 32);
    return v;
}

__device__ __forceinline__ v16h cat16(v4u lo, v4u hi) {
    v8h l = __builtin_bit_cast(v8h, lo);
    v8h h = __builtin_bit_cast(v8h, hi);
    return __builtin_shufflevector(l, h, 0,1,2,3,4,5,6,7,8,9,10,11,12,13,14,15);
}

struct Q8 { v4u q0, q1, q2, q3, q4, q5, q6, q7; };

// Issue 8 ds_load_b128 (4 K-blocks x 32B) for one 16-column B tile; no wait.
__device__ __forceinline__ void lds_issue(unsigned addr, Q8& q) {
    asm volatile(
        "ds_load_b128 %0, %8\n\t"
        "ds_load_b128 %1, %8 offset:16\n\t"
        "ds_load_b128 %2, %8 offset:64\n\t"
        "ds_load_b128 %3, %8 offset:80\n\t"
        "ds_load_b128 %4, %8 offset:128\n\t"
        "ds_load_b128 %5, %8 offset:144\n\t"
        "ds_load_b128 %6, %8 offset:192\n\t"
        "ds_load_b128 %7, %8 offset:208"
        : "=v"(q.q0), "=v"(q.q1), "=v"(q.q2), "=v"(q.q3),
          "=v"(q.q4), "=v"(q.q5), "=v"(q.q6), "=v"(q.q7)
        : "v"(addr));
}

// Wait for cur's DS loads AND immediately issue nxt's loads in the same asm
// block. Binding cur as "+v" forces every consumer (the WMMA chain) after this
// block, so nxt's loads are guaranteed to be in flight across the WMMAs.
__device__ __forceinline__ void lds_wait_issue(Q8& cur, unsigned addr, Q8& nxt) {
    asm volatile(
        "s_wait_dscnt 0x0\n\t"
        "ds_load_b128 %0, %16\n\t"
        "ds_load_b128 %1, %16 offset:16\n\t"
        "ds_load_b128 %2, %16 offset:64\n\t"
        "ds_load_b128 %3, %16 offset:80\n\t"
        "ds_load_b128 %4, %16 offset:128\n\t"
        "ds_load_b128 %5, %16 offset:144\n\t"
        "ds_load_b128 %6, %16 offset:192\n\t"
        "ds_load_b128 %7, %16 offset:208"
        : "=v"(nxt.q0), "=v"(nxt.q1), "=v"(nxt.q2), "=v"(nxt.q3),
          "=v"(nxt.q4), "=v"(nxt.q5), "=v"(nxt.q6), "=v"(nxt.q7),
          "+v"(cur.q0), "+v"(cur.q1), "+v"(cur.q2), "+v"(cur.q3),
          "+v"(cur.q4), "+v"(cur.q5), "+v"(cur.q6), "+v"(cur.q7)
        : "v"(addr));
}

// Wait only (tail tile of a panel).
__device__ __forceinline__ void lds_wait(Q8& q) {
    asm volatile("s_wait_dscnt 0x0"
        : "+v"(q.q0), "+v"(q.q1), "+v"(q.q2), "+v"(q.q3),
          "+v"(q.q4), "+v"(q.q5), "+v"(q.q6), "+v"(q.q7));
}

// One wave per row-pair i: normalize z1_i and z2_i, emit f16 rows into z_all
// (row i and row B+i), and pos_sim[i] in fp32.
__global__ void __launch_bounds__(256)
normalize_pack_kernel(const float* __restrict__ z1, const float* __restrict__ z2,
                      _Float16* __restrict__ zh, float* __restrict__ pos, int B) {
    const int wavesPerBlock = blockDim.x >> 5;
    const int w    = threadIdx.x >> 5;
    const int lane = threadIdx.x & 31;
    const int i    = blockIdx.x * wavesPerBlock + w;
    if (i >= B) return;

    const float4 f1 = ((const float4*)(z1 + (size_t)i * DDIM))[lane];
    const float4 f2 = ((const float4*)(z2 + (size_t)i * DDIM))[lane];

    float ss1 = f1.x*f1.x + f1.y*f1.y + f1.z*f1.z + f1.w*f1.w;
    float ss2 = f2.x*f2.x + f2.y*f2.y + f2.z*f2.z + f2.w*f2.w;
    float d12 = f1.x*f2.x + f1.y*f2.y + f1.z*f2.z + f1.w*f2.w;
    ss1 = waveReduceSum32(ss1);
    ss2 = waveReduceSum32(ss2);
    d12 = waveReduceSum32(d12);

    const float s1 = 1.0f / fmaxf(sqrtf(ss1), 1e-12f);
    const float s2 = 1.0f / fmaxf(sqrtf(ss2), 1e-12f);

    v4h h1, h2;
    h1.x = (_Float16)(f1.x * s1); h1.y = (_Float16)(f1.y * s1);
    h1.z = (_Float16)(f1.z * s1); h1.w = (_Float16)(f1.w * s1);
    h2.x = (_Float16)(f2.x * s2); h2.y = (_Float16)(f2.y * s2);
    h2.z = (_Float16)(f2.z * s2); h2.w = (_Float16)(f2.w * s2);

    ((v4h*)(zh + (size_t)i * DDIM))[lane]       = h1;
    ((v4h*)(zh + ((size_t)i + B) * DDIM))[lane] = h2;
    if (lane == 0) pos[i] = d12 * s1 * s2;
}

// Block = 8 waves; each wave owns a 16-row strip (block covers 128 rows).
// B panels staged in LDS by the Tensor Data Mover (double buffered, TENSORcnt
// synchronized); consumed via inline-asm ds_load_b128 pipelined against the
// v_wmma_f32_16x16x32_f16 chain, with a branchless masked row-max.
__global__ void __launch_bounds__(256)
rowmax_wmma_tdm_kernel(const _Float16* __restrict__ zh, float* __restrict__ rowmax,
                       int N, int B) {
    __shared__ _Float16 panel[2][PANEL_COLS * COL_STRIDE_H];   // sole LDS obj -> offset 0

    // Keep the LDS allocation alive (predicate never true at runtime, but the
    // compiler cannot prove it, so the reference survives).
    if (B == 0) panel[0][0] = (_Float16)1;

    const int lane    = threadIdx.x & 31;
    const int wv      = threadIdx.x >> 5;
    const int rowBase = (blockIdx.x * 8 + wv) * 16;

    // ---- A tiles (16 rows x 128 K) resident in VGPRs for the whole sweep ----
    // A 16x32 layout (matches B): lanes 0-15 (M=lane) hold K 0..15,
    // lanes 16-31 (M=lane-16) hold K 16..31.
    const int aRow     = rowBase + (lane & 15);
    const int chunkSel = (lane >= 16) ? 16 : 0;
    v16h a[4];
#pragma unroll
    for (int kb = 0; kb < 4; ++kb) {
        const _Float16* base = zh + (size_t)aRow * DDIM + kb * 32;
        v8h lo = *(const v8h*)(base + chunkSel);
        v8h hi = *(const v8h*)(base + 8 + chunkSel);
        a[kb] = __builtin_shufflevector(lo, hi, 0,1,2,3,4,5,6,7,8,9,10,11,12,13,14,15);
    }

    // B layout (32x16 K x N): col = lane&15; lanes 0-15 hold K 0..15,
    // lanes 16-31 hold K 16..31 -> 32 contiguous bytes per lane per K-block.
    const int nLane = lane & 15;
    const int koff  = (lane >= 16) ? 16 : 0;
    const int hiAdd = (lane >= 16) ? 8 : 0;    // C layout: lanes>=16 hold M=r+8
    const unsigned laneColByte = (unsigned)(nLane * COL_STRIDE_B + koff * 2);

    // Loop-invariant per-r lane predicate for the masked (i, i±B) entries.
    bool selr[8];
#pragma unroll
    for (int r = 0; r < 8; ++r) selr[r] = (nLane == r + hiAdd);

    // The masked entries of this strip fall in exactly one aligned 16-col tile.
    const int scol = (rowBase < B) ? (rowBase + B) : (rowBase - B);

    float rmax[8];
#pragma unroll
    for (int r = 0; r < 8; ++r) rmax[r] = -__builtin_inff();

    // ---- TDM descriptor issue (wave 0 only; TDM ignores EXEC) ----
    auto issue_panel_load = [&](int p, int bufIdx) {
        const unsigned long long ga =
            (unsigned long long)(size_t)(zh + (size_t)p * PANEL_COLS * DDIM);
        v4u g0;
        g0[0] = 1u;                                   // count=1, user mode
        g0[1] = (unsigned)(bufIdx * PANEL_BYTES);     // lds_addr (panel at LDS offset 0)
        g0[2] = (unsigned)(ga & 0xffffffffu);         // global_addr[31:0]
        g0[3] = (unsigned)((ga >> 32) & 0x01ffffffu)  // global_addr[56:32]
                | 0x80000000u;                        // type=2 ("image")
        v8i g1;
        g1[0] = (int)0x07510000u;   // data_size=2B | pad_enable | pad_interval=64dw | pad_amount=4dw
        g1[1] = (int)(128u << 16);                    // tensor_dim0 = 128 (lo16)
        g1[2] = (int)(((unsigned)N & 0xffffu) << 16); // tensor_dim1 lo16
        g1[3] = (int)(((unsigned)N >> 16) & 0xffffu)  // tensor_dim1 hi16
                | (int)(128u << 16);                  // tile_dim0 = 128
        g1[4] = PANEL_COLS;                           // tile_dim1 = 64, tile_dim2 = 0
        g1[5] = 128;                                  // tensor_dim0_stride = 128
        g1[6] = 0;
        g1[7] = 0;
        v4i gz = {0, 0, 0, 0};                        // 2-D: groups 2/3 unused
        v8i z8 = {0, 0, 0, 0, 0, 0, 0, 0};            // extra group (6-arg variant): unused
        __builtin_amdgcn_tensor_load_to_lds(g0, g1, gz, gz, z8, 0);
    };

    // Prologue: stage panel 0.
    if (wv == 0) {
        issue_panel_load(0, 0);
        __builtin_amdgcn_s_wait_tensorcnt(0);
    }
    __syncthreads();

    const int nPanels = N / PANEL_COLS;
    for (int p = 0; p < nPanels; ++p) {
        const int buf = p & 1;
        // Kick the next panel into the other buffer (everyone finished reading
        // it before the barrier that ended the previous iteration).
        if (wv == 0 && p + 1 < nPanels) issue_panel_load(p + 1, buf ^ 1);

        const unsigned panelBase = (unsigned)(buf * PANEL_BYTES) + laneColByte;

        // Software pipeline over the 4 tiles of this panel: tile t+1's LDS
        // loads are issued inside the same asm block that retires tile t's
        // wait, so they stay in flight across tile t's WMMA chain.
        Q8 qa, qb;
        lds_issue(panelBase, qa);
#pragma unroll
        for (int t = 0; t < PANEL_COLS / 16; ++t) {
            Q8& cur = (t & 1) ? qb : qa;
            Q8& nxt = (t & 1) ? qa : qb;
            if (t + 1 < PANEL_COLS / 16)
                lds_wait_issue(cur, panelBase + (unsigned)((t + 1) * 16 * COL_STRIDE_B), nxt);
            else
                lds_wait(cur);

            v16h b0 = cat16(cur.q0, cur.q1);
            v16h b1 = cat16(cur.q2, cur.q3);
            v16h b2 = cat16(cur.q4, cur.q5);
            v16h b3 = cat16(cur.q6, cur.q7);

            v8f c = {};
            c = __builtin_amdgcn_wmma_f32_16x16x32_f16(false, a[0], false, b0, (short)0, c, false, false);
            c = __builtin_amdgcn_wmma_f32_16x16x32_f16(false, a[1], false, b1, (short)0, c, false, false);
            c = __builtin_amdgcn_wmma_f32_16x16x32_f16(false, a[2], false, b2, (short)0, c, false, false);
            c = __builtin_amdgcn_wmma_f32_16x16x32_f16(false, a[3], false, b3, (short)0, c, false, false);

            const bool tileMasked = ((p * PANEL_COLS + t * 16) == scol);
#pragma unroll
            for (int r = 0; r < 8; ++r) {
                const float v = (tileMasked && selr[r]) ? -__builtin_inff() : c[r];
                rmax[r] = fmaxf(rmax[r], v);
            }
        }

        if (wv == 0 && p + 1 < nPanels) __builtin_amdgcn_s_wait_tensorcnt(0);
        __syncthreads();
    }

    // Reduce max across each 16-lane group (xor 1,2,4,8 stays within a group).
#pragma unroll
    for (int r = 0; r < 8; ++r) {
#pragma unroll
        for (int m = 8; m >= 1; m >>= 1)
            rmax[r] = fmaxf(rmax[r], __shfl_xor(rmax[r], m, 32));
    }
    if ((lane & 15) == 0) {
#pragma unroll
        for (int r = 0; r < 8; ++r)
            rowmax[rowBase + r + hiAdd] = rmax[r] * TEMP_INV;
    }
}

// Final scalar: mean over 2B rows of relu(hardest + margin - pos).
__global__ void __launch_bounds__(256)
loss_reduce_kernel(const float* __restrict__ rowmax, const float* __restrict__ pos,
                   float* __restrict__ out, int N, int B) {
    __shared__ float smem[256];
    float acc = 0.0f;
    for (int j = threadIdx.x; j < N; j += blockDim.x) {
        const float p = pos[(j < B) ? j : (j - B)];
        acc += fmaxf(rowmax[j] + MARGIN_C - p, 0.0f);
    }
    smem[threadIdx.x] = acc;
    __syncthreads();
    for (int s = blockDim.x >> 1; s > 0; s >>= 1) {
        if ((int)threadIdx.x < s) smem[threadIdx.x] += smem[threadIdx.x + s];
        __syncthreads();
    }
    if (threadIdx.x == 0) out[0] = smem[0] / (float)N;
}

extern "C" void kernel_launch(void* const* d_in, const int* in_sizes, int n_in,
                              void* d_out, int out_size, void* d_ws, size_t ws_size,
                              hipStream_t stream) {
    (void)n_in; (void)out_size; (void)ws_size;
    const float* z1 = (const float*)d_in[0];
    const float* z2 = (const float*)d_in[1];
    const int B = in_sizes[0] / DDIM;   // 8192
    const int N = 2 * B;                // 16384

    char* ws = (char*)d_ws;
    _Float16* zh     = (_Float16*)ws;                                        // 4 MB
    float*    rowmax = (float*)(ws + (size_t)N * DDIM * sizeof(_Float16));   // 64 KB
    float*    pos    = (float*)(ws + (size_t)N * DDIM * sizeof(_Float16)
                                   + (size_t)N * sizeof(float));             // 32 KB

    // 1) normalize + pack to f16 (one wave per row pair; 8 waves/block)
    normalize_pack_kernel<<<(B + 7) / 8, 256, 0, stream>>>(z1, z2, zh, pos, B);

    // 2) TDM-staged WMMA row-max sweep: N/16 = 1024 strips, 8 strips/block
    const int strips = N / 16;
    rowmax_wmma_tdm_kernel<<<strips / 8, 256, 0, stream>>>(zh, rowmax, N, B);

    // 3) scalar loss
    loss_reduce_kernel<<<1, 256, 0, stream>>>(rowmax, pos, (float*)d_out, N, B);
}
